// TensorGCNConv_53403623358639
// MI455X (gfx1250) — matrix operand order repaired and checked by reading, not compiled, not run on hardware
//
#include <hip/hip_runtime.h>
#include <hip/hip_bf16.h>

// out = Asum @ (x @ Wsum) + bias   (exact collapse of the C=4 FFT t-product)
//   Asum[i,j] = sum_c adjs[i,j,c], fused into GEMM2's A-tile loader
//   Wsum[j,r] = sum_c weight[j,r,c]
// GEMMs run as bf16 WMMA (v_wmma_f32_16x16x32_bf16) with f32 accumulation.

#define N_NODES 4096
#define F_IN    512
#define F_OUT   256
#define NCH     4

typedef __attribute__((ext_vector_type(16))) __bf16 v16bf;
typedef __attribute__((ext_vector_type(8)))  float  v8f;

union Frag {
  v16bf v;
  unsigned short u16[16];
  unsigned int   u32[8];
  uint4          q[2];
};

static __device__ __forceinline__ unsigned short f32_to_bf16(float f) {
  unsigned int u = __float_as_uint(f);
  u += 0x7FFFu + ((u >> 16) & 1u);   // round-to-nearest-even
  return (unsigned short)(u >> 16);
}

// Packed "B-operand order": fragment for (K-block of 32, N-block of 16) is
// 32 lanes x 16 bf16 contiguous, in the contiguous k order this code also uses for A:
//   lane = (K%32 >= 16 ? 16 : 0) + (N%16),  element e = K%16
static __device__ __forceinline__ size_t pack_idx(int k, int n) {
  int kb32 = k >> 5, kkk = k & 31;
  int laneP = (kkk & 16) + (n & 15);
  int e = kkk & 15;
  int nb = n >> 4;                       // N fixed at 256 -> 16 blocks
  return (((size_t)kb32 * 16 + nb) * 32 + laneP) * 16 + e;
}

__global__ __launch_bounds__(256)
void prep_x_kernel(const float* __restrict__ x, unsigned short* __restrict__ xb) {
  size_t t = (size_t)blockIdx.x * blockDim.x + threadIdx.x;   // 8 elems/thread
  const float4* src = (const float4*)x + t * 2;
  float4 a = src[0], b = src[1];
  uint4 o;
  o.x = (unsigned)f32_to_bf16(a.x) | ((unsigned)f32_to_bf16(a.y) << 16);
  o.y = (unsigned)f32_to_bf16(a.z) | ((unsigned)f32_to_bf16(a.w) << 16);
  o.z = (unsigned)f32_to_bf16(b.x) | ((unsigned)f32_to_bf16(b.y) << 16);
  o.w = (unsigned)f32_to_bf16(b.z) | ((unsigned)f32_to_bf16(b.w) << 16);
  ((uint4*)xb)[t] = o;
}

__global__ __launch_bounds__(256)
void prep_w_kernel(const float* __restrict__ w, unsigned short* __restrict__ wp) {
  int t = blockIdx.x * 256 + threadIdx.x;   // t in [0, F_IN*F_OUT)
  int j = t >> 8;                           // K index (F_IN)
  int r = t & 255;                          // N index (F_OUT)
  float4 c = ((const float4*)w)[t];         // weight[j][r][0..3], C fastest
  wp[pack_idx(j, r)] = f32_to_bf16(c.x + c.y + c.z + c.w);
}

// One workgroup computes a 32(M) x 256(N) output slab; 8 waves, each wave owns
// 32 columns (2x2 tiles of 16x16). BK=64 per LDS stage.
// AMODE 0: A = bf16 row-major [M,K]        (x_bf16)
// AMODE 1: A = f32 [M,K,4], sum 4 channels (adjs -> Asum, fused)
// EPI   0: store D as bf16 in packed-B order (H for GEMM2)
// EPI   1: out[i,r,0..3] = D + bias[i,r,0..3] (f32, bias broadcast fused)
template <int AMODE, int EPI>
__global__ __launch_bounds__(256)
void gemm_kernel(const void* __restrict__ Aptr,
                 const unsigned short* __restrict__ Bpack,
                 const float* __restrict__ bias,
                 float* __restrict__ out,
                 unsigned short* __restrict__ outPack,
                 int K) {
  constexpr int BM = 32, BK = 64, AP = BK + 2;   // +2 bf16 pad vs bank conflicts
  __shared__ unsigned short a_lds[BM * AP];

  const int t     = threadIdx.x;
  const int lane  = t & 31;
  const int wave  = t >> 5;
  const int Mtile = blockIdx.x * BM;
  const int half  = lane >> 4;       // which 16-lane half
  const int nlo   = lane & 15;
  const int kbase = half * 8;        // A operand: K offset per lane-half is 2*kbase (= 16*half)

  v8f acc[2][2] = {};

  for (int k0 = 0; k0 < K; k0 += BK) {
    // ---- stage A tile (BM x BK bf16) into LDS ----
    if (AMODE == 1) {
      const float4* adj4 = (const float4*)Aptr;   // one float4 per (i,j)
      const int kp = t & 31;                      // k-pair index
      const int mb = t >> 5;
#pragma unroll
      for (int mi = 0; mi < 4; ++mi) {
        const int m = mb + mi * 8;
        const float4* src = adj4 + ((size_t)(Mtile + m) * K + (k0 + 2 * kp));
        float4 c0 = src[0], c1 = src[1];
        if (k0 + BK < K) __builtin_prefetch((const void*)(src + BK), 0, 0);
        float s0 = c0.x + c0.y + c0.z + c0.w;
        float s1 = c1.x + c1.y + c1.z + c1.w;
        *(unsigned int*)&a_lds[m * AP + 2 * kp] =
            (unsigned)f32_to_bf16(s0) | ((unsigned)f32_to_bf16(s1) << 16);
      }
    } else {
      const unsigned short* xb = (const unsigned short*)Aptr;
      const int m = t >> 3, kk8 = (t & 7) << 3;
      uint4 d = *(const uint4*)(xb + (size_t)(Mtile + m) * K + k0 + kk8);
      unsigned int* dst = (unsigned int*)&a_lds[m * AP + kk8];
      dst[0] = d.x; dst[1] = d.y; dst[2] = d.z; dst[3] = d.w;
    }
    __syncthreads();

#pragma unroll
    for (int kk = 0; kk < 2; ++kk) {
      const int kb32 = (k0 >> 5) + kk;
      // B fragments straight from global (H/Wsum are L2-resident, pre-swizzled)
      Frag bfrag[2];
#pragma unroll
      for (int nt = 0; nt < 2; ++nt) {
        const uint4* src = (const uint4*)(
            Bpack + (((size_t)kb32 * 16 + (wave * 2 + nt)) * 32 + lane) * 16);
        bfrag[nt].q[0] = src[0];
        bfrag[nt].q[1] = src[1];
      }
#pragma unroll
      for (int mt = 0; mt < 2; ++mt) {
        Frag afrag;
        const int row = mt * 16 + nlo;   // A: lane holds row M = lane%16
#pragma unroll
        for (int p = 0; p < 8; ++p) {
          const int Kv = (kbase << 1) + (p << 1);
          afrag.u32[p] = *(const unsigned int*)&a_lds[row * AP + kk * 32 + Kv];
        }
#pragma unroll
        for (int nt = 0; nt < 2; ++nt) {
          acc[mt][nt] = __builtin_amdgcn_wmma_f32_16x16x32_bf16(
              false, afrag.v, false, bfrag[nt].v, (short)0, acc[mt][nt],
              false, false);
        }
      }
    }
    __syncthreads();
  }

  // ---- epilogue ----
#pragma unroll
  for (int mt = 0; mt < 2; ++mt) {
#pragma unroll
    for (int nt = 0; nt < 2; ++nt) {
      const int n = wave * 32 + nt * 16 + nlo;
#pragma unroll
      for (int v = 0; v < 8; ++v) {
        const int i = Mtile + mt * 16 + half * 8 + v;   // C/D: M = v + 8*half
        const float y = acc[mt][nt][v];
        if (EPI == 0) {
          outPack[pack_idx(i, n)] = f32_to_bf16(y);
        } else {
          const size_t base = ((size_t)i * F_OUT + n) * NCH;
          float4 bb = *(const float4*)(bias + base);
          float4 o;
          o.x = y + bb.x; o.y = y + bb.y; o.z = y + bb.z; o.w = y + bb.w;
          *(float4*)(out + base) = o;
        }
      }
    }
  }
}

extern "C" void kernel_launch(void* const* d_in, const int* in_sizes, int n_in,
                              void* d_out, int out_size, void* d_ws, size_t ws_size,
                              hipStream_t stream) {
  const float* x    = (const float*)d_in[0];   // [4096, 512]
  const float* adjs = (const float*)d_in[1];   // [4096, 4096, 4]
  const float* w    = (const float*)d_in[2];   // [512, 256, 4]
  const float* bias = (const float*)d_in[3];   // [4096, 256, 4]
  float* out = (float*)d_out;                  // [4096, 256, 4]

  char* ws = (char*)d_ws;
  unsigned short* xb    = (unsigned short*)ws;                              // 4 MB
  unsigned short* wpack = (unsigned short*)(ws + (size_t)N_NODES * F_IN * 2);      // 256 KB
  unsigned short* hpack = (unsigned short*)(ws + (size_t)N_NODES * F_IN * 2
                                               + (size_t)F_IN * F_OUT * 2);        // 2 MB

  prep_x_kernel<<<(N_NODES * F_IN / 8) / 256, 256, 0, stream>>>(x, xb);
  prep_w_kernel<<<(F_IN * F_OUT) / 256, 256, 0, stream>>>(w, wpack);
  // H = x @ Wsum  -> packed bf16
  gemm_kernel<0, 0><<<N_NODES / 32, 256, 0, stream>>>(xb, wpack, nullptr,
                                                      nullptr, hpack, F_IN);
  // out = Asum @ H + bias   (channel-sum of adjs fused into A loader)
  gemm_kernel<1, 1><<<N_NODES / 32, 256, 0, stream>>>(adjs, hpack, bias, out,
                                                      nullptr, N_NODES);
}
